// LIIF_11854109737364
// MI455X (gfx1250) — compile-verified, matrix operand order checked by reading, not run on hardware
//
#include <hip/hip_runtime.h>
#include <math.h>

// ---------------------------------------------------------------------------
// LIIF on MI455X (gfx1250, wave32).
// SIREN MLP = 262144 rows x (578*256 + 2*256*256 + 256*16) MACs ~ 147 GFLOP
// -> bf16 WMMA (v_wmma_f32_16x16x32_bf16), f32 accumulate. Encoder convs
// (6.1 GFLOP) use the same implicit-GEMM WMMA path.
// K-loops are fully unrolled via templates so all fragment loads become
// base + immediate-offset (ds_load_b128 / global_load_b128 with offset:),
// letting the scheduler overlap L2 latency with the WMMA pipe instead of
// emitting s_wait_loadcnt 0 before every WMMA.
// SIREN block = 64 MLP rows, ~144KB LDS -> 2 blocks/WGP (320KB LDS) ->
// 4 waves/SIMD for latency hiding. Weights (~1.3MB bf16) stay hot in 192MB L2.
// ---------------------------------------------------------------------------

typedef __attribute__((ext_vector_type(16))) __bf16        bf16x16;
typedef __attribute__((ext_vector_type(8)))  float         f32x8;
typedef __attribute__((ext_vector_type(4)))  unsigned int  u32x4;

union FragB16 { bf16x16 v; u32x4 q[2]; };

__device__ __forceinline__ unsigned short f2bf(float f) {
  union { float f; unsigned int u; } v; v.f = f;
  unsigned int u = v.u;
  u = u + 0x7FFFu + ((u >> 16) & 1u);   // round-to-nearest-even
  return (unsigned short)(u >> 16);
}

// 16-bit fragment: lane L holds line L%16, k-chunks at k0+8*(L/16)..+7
// (regs 0-3) and +16..+23 (regs 4-7). p points at element (line, k0+8*(L/16)).
__device__ __forceinline__ bf16x16 load_frag16(const unsigned short* p) {
  FragB16 f;
  f.q[0] = *(const u32x4*)(p);        // k .. k+7   (b128)
  f.q[1] = *(const u32x4*)(p + 16);   // k+16..k+23 (b128)
  return f.v;
}

__device__ __forceinline__ f32x8 wmma_bf16(bf16x16 a, bf16x16 b, f32x8 c) {
  return __builtin_amdgcn_wmma_f32_16x16x32_bf16(
      /*neg_a=*/false, a, /*neg_b=*/false, b,
      /*c_mod=*/(short)0, c, /*reuse_a=*/false, /*reuse_b=*/false);
}

__device__ __forceinline__ void cubicw(float t, float* w) {
  const float a = -0.75f;
  float t1 = t + 1.f;
  w[0] = ((a * t1 - 5.f * a) * t1 + 8.f * a) * t1 - 4.f * a;
  w[1] = ((a + 2.f) * t - (a + 3.f)) * t * t + 1.f;
  float s = 1.f - t;
  w[2] = ((a + 2.f) * s - (a + 3.f)) * s * s + 1.f;
  float s1 = s + 1.f;
  w[3] = ((a * s1 - 5.f * a) * s1 + 8.f * a) * s1 - 4.f * a;
}

// ---------------------------------------------------------------------------
// Weight packing: fp32 [Ns][Ks] -> bf16 [Nd][Kd], zero padded.
// ---------------------------------------------------------------------------
__global__ void pack_bf16_kernel(const float* __restrict__ src,
                                 unsigned short* __restrict__ dst,
                                 int Ns, int Ks, int Nd, int Kd) {
  int i = blockIdx.x * blockDim.x + threadIdx.x;
  int tot = Nd * Kd;
  if (i >= tot) return;
  int nI = i / Kd;
  int k  = i - nI * Kd;
  dst[i] = (nI < Ns && k < Ks) ? f2bf(src[(size_t)nI * Ks + k]) : (unsigned short)0;
}

// ---------------------------------------------------------------------------
// conv3x3 (pad 1) as implicit GEMM: M = 64 pixels/block, N = 64 out channels,
// K = Cin*9 padded to KP (template -> fully unrolled k-loop, immediate-offset
// fragment loads). 4 waves: wave w owns M-tile w, all 4 N-tiles.
// Fused bias + optional ReLU + optional residual add. fp32 NCHW activations.
// ---------------------------------------------------------------------------
template <int KP>
__global__ void __launch_bounds__(128)
conv3x3_wmma_kernel(const float* __restrict__ in,
                    const float* __restrict__ skip,
                    float* __restrict__ out,
                    const unsigned short* __restrict__ Wp,  // [64][KP] bf16
                    const float* __restrict__ bias,         // [64]
                    int Cin, int do_relu, int do_skip) {
  extern __shared__ __align__(16) unsigned char smemRaw[];
  unsigned short* sm = (unsigned short*)smemRaw;   // [64][KP+8]
  constexpr int AST = KP + 8;
  const int tid = threadIdx.x;
  const int wgBase = blockIdx.x * 64;              // first pixel of this block

  // ---- im2col gather -> bf16 LDS ----
  {
    int r   = tid >> 1;
    int par = tid & 1;
    int pixel = wgBase + r;
    int b   = pixel / 2304;
    int rem = pixel - b * 2304;
    int y   = rem / 48;
    int x   = rem - y * 48;
    const float* inB = in + (size_t)b * Cin * 2304;
    const int KV = Cin * 9;
    for (int k = par; k < AST; k += 2) {
      float v = 0.f;
      if (k < KV) {
        int ci = k / 9;
        int jk = k - ci * 9;
        int dj = jk / 3;
        int dk = jk - dj * 3;
        int yy = y + dj - 1, xx = x + dk - 1;
        if (yy >= 0 && yy < 48 && xx >= 0 && xx < 48)
          v = inB[(size_t)ci * 2304 + yy * 48 + xx];
      }
      sm[r * AST + k] = f2bf(v);
    }
  }
  __syncthreads();

  const int wave = tid >> 5;
  const int lane = tid & 31;
  const int hh   = lane >> 4;     // lane half
  const int nl   = lane & 15;

  f32x8 acc[4];
#pragma unroll
  for (int nt = 0; nt < 4; ++nt)
#pragma unroll
    for (int i = 0; i < 8; ++i) acc[nt][i] = 0.f;

  const unsigned short* ap = sm + (wave * 16 + nl) * AST + hh * 8;
  const unsigned short* bp0 = Wp + (size_t)(0 * 16 + nl) * KP + hh * 8;
  const unsigned short* bp1 = Wp + (size_t)(1 * 16 + nl) * KP + hh * 8;
  const unsigned short* bp2 = Wp + (size_t)(2 * 16 + nl) * KP + hh * 8;
  const unsigned short* bp3 = Wp + (size_t)(3 * 16 + nl) * KP + hh * 8;

#pragma unroll
  for (int kt = 0; kt < KP / 32; ++kt) {
    const int kb = kt * 32;
    bf16x16 af = load_frag16(ap + kb);
    acc[0] = wmma_bf16(af, load_frag16(bp0 + kb), acc[0]);
    acc[1] = wmma_bf16(af, load_frag16(bp1 + kb), acc[1]);
    acc[2] = wmma_bf16(af, load_frag16(bp2 + kb), acc[2]);
    acc[3] = wmma_bf16(af, load_frag16(bp3 + kb), acc[3]);
  }

  // ---- epilogue: bias (+skip)(+relu) -> fp32 NCHW ----
#pragma unroll
  for (int nt = 0; nt < 4; ++nt) {
    int n = nt * 16 + nl;
    float bb = bias[n];
#pragma unroll
    for (int r = 0; r < 8; ++r) {
      int m = wave * 16 + r + hh * 8;        // D layout: reg r -> row r + 8*half
      int pixel = wgBase + m;
      int b   = pixel / 2304;
      int rem = pixel - b * 2304;
      size_t idx = ((size_t)(b * 64 + n)) * 2304 + rem;
      float v = acc[nt][r] + bb;
      if (do_skip) v += skip[idx];
      if (do_relu) v = fmaxf(v, 0.f);
      out[idx] = v;
    }
  }
}

// ---------------------------------------------------------------------------
// SIREN layer: sOut[64x256] = sin(30*(sIn[64xK] * W^T + b)), bf16 WMMA.
// 8 waves: wave w owns N-tiles {2w, 2w+1} x all 4 M-tiles (acc = 64 VGPRs).
// K templated -> fully unrolled, immediate-offset fragment loads.
// ---------------------------------------------------------------------------
template <int K, int IST, int OST>
__device__ __forceinline__ void mlp_layer_sin(
    const unsigned short* __restrict__ sIn,
    unsigned short* __restrict__ sOut,
    const unsigned short* __restrict__ Wg,   // [256][K] bf16
    const float* __restrict__ bias,          // [256]
    int wave, int lane) {
  const int hh = lane >> 4, nl = lane & 15;
  f32x8 acc[4][2];
#pragma unroll
  for (int mt = 0; mt < 4; ++mt)
#pragma unroll
    for (int t = 0; t < 2; ++t)
#pragma unroll
      for (int i = 0; i < 8; ++i) acc[mt][t][i] = 0.f;

  const int n0 = (wave * 2 + 0) * 16 + nl;
  const int n1 = (wave * 2 + 1) * 16 + nl;
  const unsigned short* bp0 = Wg + (size_t)n0 * K + hh * 8;
  const unsigned short* bp1 = Wg + (size_t)n1 * K + hh * 8;
  const unsigned short* ap  = sIn + nl * IST + hh * 8;

#pragma unroll
  for (int kt = 0; kt < K / 32; ++kt) {
    const int kb = kt * 32;
    bf16x16 bf0 = load_frag16(bp0 + kb);
    bf16x16 bf1 = load_frag16(bp1 + kb);
#pragma unroll
    for (int mt = 0; mt < 4; ++mt) {
      bf16x16 af = load_frag16(ap + mt * 16 * IST + kb);
      acc[mt][0] = wmma_bf16(af, bf0, acc[mt][0]);
      acc[mt][1] = wmma_bf16(af, bf1, acc[mt][1]);
    }
  }
#pragma unroll
  for (int t = 0; t < 2; ++t) {
    int n = (wave * 2 + t) * 16 + nl;
    float bb = bias[n];
#pragma unroll
    for (int mt = 0; mt < 4; ++mt)
#pragma unroll
      for (int r = 0; r < 8; ++r) {
        int m = mt * 16 + r + hh * 8;
        sOut[m * OST + n] = f2bf(__sinf(30.f * (acc[mt][t][r] + bb)));
      }
  }
}

// ---------------------------------------------------------------------------
// LIIF mega-kernel: block = 16 query points x 4 shifts = 64 MLP rows, 8 waves.
// Phase 1: bicubic gather of unfolded features (+rel) -> LDS bf16 [64][616].
// Phases 2-4: 3x WMMA sin-layers + 256->16 output tile + sigmoid + blend.
// ---------------------------------------------------------------------------
__global__ void __launch_bounds__(256)
liif_siren_kernel(const float* __restrict__ feats,        // [4][64][48][48]
                  const float* __restrict__ coords,       // [4][16384][2]
                  const unsigned short* __restrict__ W0,  // [256][608]
                  const float* __restrict__ b0,           // [256]
                  const unsigned short* __restrict__ W1,  // [2][256][256]
                  const float* __restrict__ b1,           // [2][256]
                  const unsigned short* __restrict__ Wo,  // [16][256]
                  const float* __restrict__ bo,           // [3]
                  float* __restrict__ outRGB)             // [4][16384][3]
{
  extern __shared__ __align__(16) unsigned char smemRaw[];
  unsigned short* inA  = (unsigned short*)smemRaw;     // 64 x 616
  unsigned short* hA   = inA + 64 * 616;               // 64 x 264
  unsigned short* hB   = hA + 64 * 264;                // 64 x 264
  float* areaS = (float*)(hB + 64 * 264);              // 64
  float* predS = areaS + 64;                           // 64 x 4

  const int tid = threadIdx.x;
  const int wg  = blockIdx.x;

  // ================= Phase 1: gather =================
  {
    const int p     = tid >> 4;     // point 0..15
    const int chunk = tid & 15;     // 4 feature channels per thread
    const int point = wg * 16 + p;
    const int b  = point >> 14;
    const int qn = point & 16383;
    const float cx = coords[((size_t)(b * 16384 + qn)) * 2 + 0];
    const float cy = coords[((size_t)(b * 16384 + qn)) * 2 + 1];
    const float eps = 1e-6f;

    for (int s = 0; s < 4; ++s) {
      const int row = p * 4 + s;
      const float dx = (s & 2) ? 1.f : -1.f;
      const float dy = (s & 1) ? 1.f : -1.f;
      float cxs = fminf(fmaxf(cx + dx * (1.f / 48.f) + eps, -1.f + eps), 1.f - eps);
      float cys = fminf(fmaxf(cy + dy * (1.f / 48.f) + eps, -1.f + eps), 1.f - eps);
      float ix = ((cxs + 1.f) * 48.f - 1.f) * 0.5f;
      float iy = ((cys + 1.f) * 48.f - 1.f) * 0.5f;
      float x0f = floorf(ix), y0f = floorf(iy);
      int x0 = (int)x0f, y0 = (int)y0f;
      float wx[4], wyv[4];
      cubicw(ix - x0f, wx);
      cubicw(iy - y0f, wyv);

      float wjk[16];                         // bicubic weights * in-range mask
#pragma unroll
      for (int j = 0; j < 4; ++j) {
        int yy = y0 - 1 + j;
        bool ym = (yy >= 0) && (yy < 48);
#pragma unroll
        for (int k = 0; k < 4; ++k) {
          int xx = x0 - 1 + k;
          bool m = ym && (xx >= 0) && (xx < 48);
          wjk[j * 4 + k] = m ? wyv[j] * wx[k] : 0.f;
        }
      }

      // features: 4 channels x 9 unfold taps. feats9[ci*9+p] is the bicubic
      // sample of the (pj,pk)-shifted zero-padded feature map -> shared 6x6
      // window per channel.
      for (int cc = 0; cc < 4; ++cc) {
        int ci = chunk * 4 + cc;
        const float* F = feats + ((size_t)(b * 64 + ci)) * 2304;
        float win[36];
#pragma unroll
        for (int jj = 0; jj < 6; ++jj) {
          int yy = y0 - 2 + jj;
#pragma unroll
          for (int kk = 0; kk < 6; ++kk) {
            int xx = x0 - 2 + kk;
            win[jj * 6 + kk] = (yy >= 0 && yy < 48 && xx >= 0 && xx < 48)
                                   ? F[yy * 48 + xx] : 0.f;
          }
        }
#pragma unroll
        for (int pq = 0; pq < 9; ++pq) {
          int pj = pq / 3, pk = pq - pj * 3;
          float sacc = 0.f;
#pragma unroll
          for (int j = 0; j < 4; ++j)
#pragma unroll
            for (int k = 0; k < 4; ++k)
              sacc += wjk[j * 4 + k] * win[(j + pj) * 6 + (k + pk)];
          inA[row * 616 + ci * 9 + pq] = f2bf(sacc);
        }
      }

      if (chunk == 0) {   // rel coords, area, zero-pad tail
        float csx = 0.f, csy = 0.f;
#pragma unroll
        for (int j = 0; j < 4; ++j) {
          int yc = min(max(y0 - 1 + j, 0), 47);
#pragma unroll
          for (int k = 0; k < 4; ++k) {
            int xc = min(max(x0 - 1 + k, 0), 47);
            float w = wjk[j * 4 + k];
            csx += w * ((2 * xc + 1) * (1.f / 48.f) - 1.f);
            csy += w * ((2 * yc + 1) * (1.f / 48.f) - 1.f);
          }
        }
        float r0 = (cx - csx) * 48.f;
        float r1 = (cy - csy) * 48.f;
        areaS[row] = fabsf(r0 * r1) + 1e-9f;
        inA[row * 616 + 576] = f2bf(r0);
        inA[row * 616 + 577] = f2bf(r1);
        for (int k = 578; k < 616; ++k) inA[row * 616 + k] = 0;
      }
    }
  }
  __syncthreads();

  // ================= Phases 2-4: SIREN WMMA =================
  const int wave = tid >> 5, lane = tid & 31;
  mlp_layer_sin<608, 616, 264>(inA, hA, W0, b0, wave, lane);
  __syncthreads();
  mlp_layer_sin<256, 264, 264>(hA, hB, W1, b1, wave, lane);
  __syncthreads();
  mlp_layer_sin<256, 264, 264>(hB, hA, W1 + 256 * 256, b1 + 256, wave, lane);
  __syncthreads();

  // output layer: waves 0-3 own one 16-row M-tile each, single N-tile
  if (wave < 4) {
    const int hh = lane >> 4, nl = lane & 15;
    f32x8 acc;
#pragma unroll
    for (int i = 0; i < 8; ++i) acc[i] = 0.f;
    const unsigned short* bp = Wo + nl * 256 + hh * 8;
    const unsigned short* ap = hA + (wave * 16 + nl) * 264 + hh * 8;
#pragma unroll
    for (int kt = 0; kt < 8; ++kt) {
      acc = wmma_bf16(load_frag16(ap + kt * 32), load_frag16(bp + kt * 32), acc);
    }
    if (nl < 3) {
      float bb = bo[nl];
#pragma unroll
      for (int r = 0; r < 8; ++r) {
        int m = wave * 16 + r + hh * 8;
        float v = acc[r] + bb;
        predS[m * 4 + nl] = 1.f / (1.f + __expf(-v));
      }
    }
  }
  __syncthreads();

  // blend: weights use reversed areas
  if (tid < 16) {
    int p = tid;
    int point = wg * 16 + p;
    float a0 = areaS[p * 4 + 0], a1 = areaS[p * 4 + 1];
    float a2 = areaS[p * 4 + 2], a3 = areaS[p * 4 + 3];
    float inv = 1.f / (a0 + a1 + a2 + a3);
    float w0 = a3 * inv, w1 = a2 * inv, w2 = a1 * inv, w3 = a0 * inv;
#pragma unroll
    for (int c = 0; c < 3; ++c) {
      float v = predS[(p * 4 + 0) * 4 + c] * w0
              + predS[(p * 4 + 1) * 4 + c] * w1
              + predS[(p * 4 + 2) * 4 + c] * w2
              + predS[(p * 4 + 3) * 4 + c] * w3;
      outRGB[(size_t)point * 3 + c] = v;
    }
  }
}

// ---------------------------------------------------------------------------
// Host launcher
// ---------------------------------------------------------------------------
extern "C" void kernel_launch(void* const* d_in, const int* in_sizes, int n_in,
                              void* d_out, int out_size, void* d_ws, size_t ws_size,
                              hipStream_t stream) {
  (void)in_sizes; (void)n_in; (void)out_size; (void)ws_size;
  const float* lrs    = (const float*)d_in[0];
  const float* coords = (const float*)d_in[1];
  // d_in[2] = cells (unused by the reference)
  const float* e0_w  = (const float*)d_in[3];
  const float* e0_b  = (const float*)d_in[4];
  const float* rb_w1 = (const float*)d_in[5];
  const float* rb_b1 = (const float*)d_in[6];
  const float* rb_w2 = (const float*)d_in[7];
  const float* rb_b2 = (const float*)d_in[8];
  const float* e1_w  = (const float*)d_in[9];
  const float* e1_b  = (const float*)d_in[10];
  const float* s0_w  = (const float*)d_in[11];
  const float* s0_b  = (const float*)d_in[12];
  const float* s1_w  = (const float*)d_in[13];
  const float* s1_b  = (const float*)d_in[14];
  const float* so_w  = (const float*)d_in[15];
  const float* so_b  = (const float*)d_in[16];
  float* out = (float*)d_out;

  // ---- workspace carve (~8.4 MB) ----
  unsigned char* ws = (unsigned char*)d_ws;
  size_t off = 0;
  auto carve = [&](size_t bytes) {
    unsigned char* p = ws + off;
    off = (off + bytes + 255) & ~(size_t)255;
    return p;
  };
  unsigned short* w0p  = (unsigned short*)carve((size_t)256 * 608 * 2);
  unsigned short* w1p  = (unsigned short*)carve((size_t)2 * 256 * 256 * 2);
  unsigned short* wop  = (unsigned short*)carve((size_t)16 * 256 * 2);
  unsigned short* e0p  = (unsigned short*)carve((size_t)64 * 32 * 2);
  unsigned short* rb1p = (unsigned short*)carve((size_t)4 * 64 * 576 * 2);
  unsigned short* rb2p = (unsigned short*)carve((size_t)4 * 64 * 576 * 2);
  unsigned short* e1p  = (unsigned short*)carve((size_t)64 * 576 * 2);
  float* actA = (float*)carve((size_t)4 * 64 * 2304 * 4);
  float* actB = (float*)carve((size_t)4 * 64 * 2304 * 4);
  float* actC = (float*)carve((size_t)4 * 64 * 2304 * 4);

  auto pack = [&](const float* src, unsigned short* dst, int Ns, int Ks, int Nd, int Kd) {
    int tot = Nd * Kd;
    pack_bf16_kernel<<<(tot + 255) / 256, 256, 0, stream>>>(src, dst, Ns, Ks, Nd, Kd);
  };
  pack(s0_w, w0p, 256, 578, 256, 608);
  pack(s1_w, w1p, 512, 256, 512, 256);
  pack(so_w, wop, 3, 256, 16, 256);
  pack(e0_w, e0p, 64, 27, 64, 32);
  pack(rb_w1, rb1p, 256, 576, 256, 576);
  pack(rb_w2, rb2p, 256, 576, 256, 576);
  pack(e1_w, e1p, 64, 576, 64, 576);

  const size_t convSmemBig = (size_t)64 * (576 + 8) * 2;                  // ~73 KB
  const size_t convSmemE0  = (size_t)64 * (32 + 8) * 2;                   // 5 KB
  const size_t sirenSmem   = (size_t)(64 * 616 + 2 * 64 * 264) * 2
                           + (size_t)(64 + 256) * 4;                      // ~144 KB
  (void)hipFuncSetAttribute(reinterpret_cast<const void*>(conv3x3_wmma_kernel<576>),
                            hipFuncAttributeMaxDynamicSharedMemorySize, (int)convSmemBig);
  (void)hipFuncSetAttribute(reinterpret_cast<const void*>(conv3x3_wmma_kernel<32>),
                            hipFuncAttributeMaxDynamicSharedMemorySize, (int)convSmemE0);
  (void)hipFuncSetAttribute(reinterpret_cast<const void*>(liif_siren_kernel),
                            hipFuncAttributeMaxDynamicSharedMemorySize, (int)sirenSmem);

  // encoder: 9216 pixels / 64 per block = 144 blocks per conv
  conv3x3_wmma_kernel<32><<<144, 128, convSmemE0, stream>>>(
      lrs, nullptr, actA, e0p, e0_b, 3, 1, 0);
  float* h = actA;
  float* r = actB;
  float* t = actC;
  for (int i = 0; i < 4; ++i) {
    conv3x3_wmma_kernel<576><<<144, 128, convSmemBig, stream>>>(
        h, nullptr, r, rb1p + (size_t)i * 64 * 576, rb_b1 + i * 64, 64, 1, 0);
    conv3x3_wmma_kernel<576><<<144, 128, convSmemBig, stream>>>(
        r, h, t, rb2p + (size_t)i * 64 * 576, rb_b2 + i * 64, 64, 0, 1);
    float* tmp = h; h = t; t = tmp;
  }
  conv3x3_wmma_kernel<576><<<144, 128, convSmemBig, stream>>>(
      h, nullptr, r, e1p, e1_b, 64, 0, 0);   // feats -> r

  // gather + SIREN + blend: 65536 points / 16 per block = 4096 blocks
  liif_siren_kernel<<<4096, 256, sirenSmem, stream>>>(
      r, coords, w0p, s0_b, w1p, s1_b, wop, so_b, out);
}
